// TransformerBlock_15195594293661
// MI455X (gfx1250) — matrix operand order repaired and checked by reading, not compiled, not run on hardware
//
#include <hip/hip_runtime.h>
#include <hip/hip_bf16.h>
#include <math.h>
#include <stdint.h>

// ---------------------------------------------------------------------------
// Types
// ---------------------------------------------------------------------------
typedef __bf16 bf16;
typedef __attribute__((ext_vector_type(8)))  __bf16 v8bf;
typedef __attribute__((ext_vector_type(16))) __bf16 v16bf;
typedef __attribute__((ext_vector_type(8)))  float  v8f;

#define WMMA_BF16(A, B, C) \
  __builtin_amdgcn_wmma_f32_16x16x32_bf16(false, (A), false, (B), (short)0, (C), false, false)

// ---------------------------------------------------------------------------
// CDNA5 async global->LDS staging (ASYNCcnt-tracked) via inline asm.
// The clang builtin exists but its pointer params live in the cuda_device
// LangAS which is not spellable from HIP source, so we emit the instruction
// directly (bridge-doc-recommended path; bypasses Sema entirely).
//   VDST  = per-lane LDS byte offset (low 32 bits of generic LDS address)
//   VADDR = 64-bit global address (VGPR pair), SADDR = off
// ---------------------------------------------------------------------------
#if defined(__gfx1250__)
#define HAVE_ASYNC_LDS 1
#else
#define HAVE_ASYNC_LDS 0
#endif

__device__ __forceinline__ void cp_async_b128(bf16* lds_dst, const bf16* gsrc) {
#if HAVE_ASYNC_LDS
  unsigned           lds_off = (unsigned)(uintptr_t)lds_dst;
  unsigned long long gaddr   = (unsigned long long)(uintptr_t)gsrc;
  asm volatile("global_load_async_to_lds_b128 %0, %1, off"
               :
               : "v"(lds_off), "v"(gaddr)
               : "memory");
#else
  *(v8bf*)lds_dst = *(const v8bf*)gsrc;
#endif
}

template <int N>
__device__ __forceinline__ void wait_async_le() {
#if HAVE_ASYNC_LDS
#if __has_builtin(__builtin_amdgcn_s_wait_asynccnt)
  __builtin_amdgcn_s_wait_asynccnt(N);
#else
  asm volatile("s_wait_asynccnt %0" ::"i"(N) : "memory");
#endif
#endif
}

__device__ __forceinline__ v16bf cat8(v8bf lo, v8bf hi) {
  union { v16bf v; v8bf h[2]; } u;
  u.h[0] = lo; u.h[1] = hi;
  return u.v;
}

// A operand fragment, 16x32 bf16 tile (row-major in LDS, K contiguous).
// ISA layout: lanes 0-15: M=lane, K={0..7,16..23}; lanes 16-31: M=lane-16, K={8..15,24..31}
__device__ __forceinline__ v16bf lds_load_a_frag(const bf16* tile, int pitch) {
  const int lane = threadIdx.x & 31;
  const int m  = lane & 15;
  const int ks = lane >> 4;
  const bf16* p = tile + m * pitch + ks * 8;
  v8bf lo = *(const v8bf*)(p);
  v8bf hi = *(const v8bf*)(p + 16);
  return cat8(lo, hi);
}

// B operand fragment for a 32x16 (KxN) MAC, sourced from a [N][K] row-major LDS
// tile. ISA layout: lane holds column N=lane&15; lanes 0-15: K={0..7,16..23}, lanes 16-31: K={8..15,24..31}.
__device__ __forceinline__ v16bf lds_load_b_frag(const bf16* tile, int pitch) {
  const int lane = threadIdx.x & 31;
  const int n  = lane & 15;
  const int ks = lane >> 4;
  const bf16* p = tile + n * pitch + ks * 8;
  v8bf lo = *(const v8bf*)(p);
  v8bf hi = *(const v8bf*)(p + 16);
  return cat8(lo, hi);
}

// ---------------------------------------------------------------------------
// fp32 -> bf16 conversion (weights)
// ---------------------------------------------------------------------------
__global__ void f32_to_bf16_kernel(const float* __restrict__ src, bf16* __restrict__ dst, int n) {
  int i = blockIdx.x * blockDim.x + threadIdx.x;
  if (i < n) dst[i] = (bf16)src[i];
}

// ---------------------------------------------------------------------------
// LayerNorm over C=1024, one block (256 threads) per row, bf16 output
// ---------------------------------------------------------------------------
__global__ __launch_bounds__(256)
void layernorm_to_bf16_kernel(const float* __restrict__ x, const float* __restrict__ g,
                              const float* __restrict__ b, bf16* __restrict__ out, int C) {
  const int row = blockIdx.x;
  const int tid = threadIdx.x;
  __shared__ float red[256];
  const float* xr = x + (size_t)row * C;

  float v[4];
  float s = 0.f;
#pragma unroll
  for (int i = 0; i < 4; ++i) { v[i] = xr[tid + i * 256]; s += v[i]; }
  red[tid] = s; __syncthreads();
  for (int st = 128; st > 0; st >>= 1) { if (tid < st) red[tid] += red[tid + st]; __syncthreads(); }
  float mu = red[0] / (float)C;
  __syncthreads();

  float s2 = 0.f;
#pragma unroll
  for (int i = 0; i < 4; ++i) { float d = v[i] - mu; s2 += d * d; }
  red[tid] = s2; __syncthreads();
  for (int st = 128; st > 0; st >>= 1) { if (tid < st) red[tid] += red[tid + st]; __syncthreads(); }
  float rstd = rsqrtf(red[0] / (float)C + 1e-5f);

#pragma unroll
  for (int i = 0; i < 4; ++i) {
    int c = tid + i * 256;
    out[(size_t)row * C + c] = (bf16)((v[i] - mu) * rstd * g[c] + b[c]);
  }
}

// ---------------------------------------------------------------------------
// Tiled bf16 GEMM with fused epilogue: D[M,N] = A[M,K] @ B[N,K]^T (+bias,...)
//   mode 0: store bf16     mode 1: +residual, store f32     mode 2: GELU, bf16
// Block tile 64x128, BK=64, double-buffered async LDS staging.
// 256 threads = 8 waves of 32x32 wave tiles; 8 WMMA / wave / stage.
// ---------------------------------------------------------------------------
#define BM 64
#define BN 128
#define BK 64
#define TP 72   // LDS row pitch (bf16): 144 B rows, 16B-aligned, conflict-free

__global__ __launch_bounds__(256)
void gemm_bf16_wmma_kernel(const bf16* __restrict__ A, const bf16* __restrict__ Bm,
                           const float* __restrict__ bias, const float* __restrict__ resid,
                           float* __restrict__ outF, bf16* __restrict__ outB,
                           int M, int N, int K, int mode) {
  __shared__ __align__(16) bf16 sA[2][BM * TP];
  __shared__ __align__(16) bf16 sB[2][BN * TP];

  const int tid  = threadIdx.x;
  const int lane = tid & 31;
  const int wave = tid >> 5;   // 0..7
  const int wm   = wave >> 2;  // 0..1  -> 32-row slice
  const int wn   = wave & 3;   // 0..3  -> 32-col slice
  const int m0   = blockIdx.y * BM;
  const int n0   = blockIdx.x * BN;

  v8f acc[2][2] = {};

  // Stage one BK=64 deep tile pair: 6 async b128 per thread (2 for A, 4 for B)
  auto stage = [&](int buf, int k0) {
#pragma unroll
    for (int i = 0; i < 2; ++i) {
      int s = tid + i * 256;
      int r = s >> 3, c = (s & 7) * 8;
      cp_async_b128(&sA[buf][r * TP + c], &A[(size_t)(m0 + r) * K + k0 + c]);
    }
#pragma unroll
    for (int i = 0; i < 4; ++i) {
      int s = tid + i * 256;
      int r = s >> 3, c = (s & 7) * 8;
      cp_async_b128(&sB[buf][r * TP + c], &Bm[(size_t)(n0 + r) * K + k0 + c]);
    }
  };

  const int nst = K / BK;
  stage(0, 0);
  for (int i = 0; i < nst; ++i) {
    const int buf = i & 1;
    if (i + 1 < nst) {
      stage(buf ^ 1, (i + 1) * BK);  // prefetch next stage under this stage's math
      wait_async_le<6>();            // this stage's 6 loads done; next 6 in flight
    } else {
      wait_async_le<0>();
    }
    __syncthreads();

#pragma unroll
    for (int kk = 0; kk < 2; ++kk) {
      const bf16* pa = &sA[buf][kk * 32];
      const bf16* pb = &sB[buf][kk * 32];
      v16bf a0 = lds_load_a_frag(pa + (wm * 32) * TP, TP);
      v16bf a1 = lds_load_a_frag(pa + (wm * 32 + 16) * TP, TP);
      v16bf b0 = lds_load_b_frag(pb + (wn * 32) * TP, TP);
      v16bf b1 = lds_load_b_frag(pb + (wn * 32 + 16) * TP, TP);
      acc[0][0] = WMMA_BF16(a0, b0, acc[0][0]);
      acc[0][1] = WMMA_BF16(a0, b1, acc[0][1]);
      acc[1][0] = WMMA_BF16(a1, b0, acc[1][0]);
      acc[1][1] = WMMA_BF16(a1, b1, acc[1][1]);
    }
    __syncthreads();
  }

  const int lm = lane & 15;
  const int lh = lane >> 4;
#pragma unroll
  for (int ti = 0; ti < 2; ++ti) {
#pragma unroll
    for (int tj = 0; tj < 2; ++tj) {
      int col = n0 + wn * 32 + tj * 16 + lm;
      float bv = bias[col];
#pragma unroll
      for (int r = 0; r < 8; ++r) {
        int row = m0 + wm * 32 + ti * 16 + r + 8 * lh;
        float v = acc[ti][tj][r] + bv;
        size_t idx = (size_t)row * N + col;
        if (mode == 0) {
          outB[idx] = (bf16)v;
        } else if (mode == 1) {
          outF[idx] = v + resid[idx];
        } else {
          float gl = 0.5f * v * (1.0f + erff(v * 0.70710678118654752f));
          outB[idx] = (bf16)gl;
        }
      }
    }
  }
}

// ---------------------------------------------------------------------------
// Causal flash attention, head_dim=64. qkv: [B*T, 3*nh*hd] bf16.
// Block = 64 query rows of one (b,h); 128 threads = 4 waves x 16 rows.
// ---------------------------------------------------------------------------
#define QP 72

__global__ __launch_bounds__(128)
void flash_attn_wmma_kernel(const bf16* __restrict__ qkv, bf16* __restrict__ attnOut,
                            int T, int nh, int hd) {
  const int C3   = 3 * nh * hd;     // 3072
  const int qblk = blockIdx.x;
  const int bh   = blockIdx.y;
  const int b    = bh / nh, h = bh % nh;
  const int tid  = threadIdx.x;
  const int lane = tid & 31;
  const int wave = tid >> 5;        // 0..3
  const int lm   = lane & 15;
  const int lh   = lane >> 4;

  __shared__ __align__(16) bf16 sQ [64 * QP];
  __shared__ __align__(16) bf16 sK [64 * QP];
  __shared__ __align__(16) bf16 sVt[64 * QP];   // V transposed: [d][key]
  __shared__ __align__(16) bf16 sP [64 * QP];   // per-wave 16-row P tiles

  const bf16* qbase = qkv + (size_t)(b * T) * C3 + h * hd;
  const bf16* kbase = qbase + nh * hd;
  const bf16* vbase = qbase + 2 * nh * hd;
  const int q0 = qblk * 64;

  // Load Q block (64 x 64 bf16) via async direct-to-LDS
  for (int i = tid; i < 64 * 8; i += 128) {
    int r = i >> 3, c = (i & 7) * 8;
    cp_async_b128(&sQ[r * QP + c], &qbase[(size_t)(q0 + r) * C3 + c]);
  }

  float mrun[8], lrun[8];
  v8f o[4] = {};
#pragma unroll
  for (int r = 0; r < 8; ++r) { mrun[r] = -INFINITY; lrun[r] = 0.f; }

  for (int j = 0; j <= qblk; ++j) {
    // Stage K row-major (async), V transposed (VGPR path: needs transpose)
    for (int i = tid; i < 64 * 8; i += 128) {
      int r = i >> 3, c = (i & 7) * 8;
      cp_async_b128(&sK[r * QP + c], &kbase[(size_t)(j * 64 + r) * C3 + c]);
    }
    for (int i = tid; i < 64 * 32; i += 128) {
      int r = i >> 5, d = (i & 31) * 2;
      const bf16* vp = &vbase[(size_t)(j * 64 + r) * C3 + d];
      sVt[d * QP + r]       = vp[0];
      sVt[(d + 1) * QP + r] = vp[1];
    }
    wait_async_le<0>();
    __syncthreads();

    // S = Q @ K^T (16 x 64 per wave), fp32 accum
    v8f s[4] = {};
#pragma unroll
    for (int kk = 0; kk < 2; ++kk) {
      v16bf aq = lds_load_a_frag(&sQ[(wave * 16) * QP + kk * 32], QP);
#pragma unroll
      for (int nt = 0; nt < 4; ++nt) {
        v16bf bk = lds_load_b_frag(&sK[(nt * 16) * QP + kk * 32], QP);
        s[nt] = WMMA_BF16(aq, bk, s[nt]);
      }
    }

    // scale 1/sqrt(64) + causal mask on diagonal block
#pragma unroll
    for (int nt = 0; nt < 4; ++nt) {
#pragma unroll
      for (int r = 0; r < 8; ++r) {
        float sv = s[nt][r] * 0.125f;
        if (j == qblk) {
          int key = j * 64 + nt * 16 + lm;
          int qr  = q0 + wave * 16 + r + 8 * lh;
          if (key > qr) sv = -INFINITY;
        }
        s[nt][r] = sv;
      }
    }

    // Online softmax: row reductions across 16 lanes (wave32 butterfly)
#pragma unroll
    for (int r = 0; r < 8; ++r) {
      float mx = fmaxf(fmaxf(s[0][r], s[1][r]), fmaxf(s[2][r], s[3][r]));
#pragma unroll
      for (int off = 1; off < 16; off <<= 1)
        mx = fmaxf(mx, __shfl_xor(mx, off, 32));
      float mnew  = fmaxf(mrun[r], mx);
      float alpha = expf(mrun[r] - mnew);
      float rs = 0.f;
#pragma unroll
      for (int nt = 0; nt < 4; ++nt) {
        float p = expf(s[nt][r] - mnew);
        s[nt][r] = p;
        rs += p;
      }
#pragma unroll
      for (int off = 1; off < 16; off <<= 1)
        rs += __shfl_xor(rs, off, 32);
      lrun[r] = lrun[r] * alpha + rs;
      mrun[r] = mnew;
#pragma unroll
      for (int nt = 0; nt < 4; ++nt) o[nt][r] *= alpha;
    }

    // Re-layout P (C-layout -> A-layout) through per-wave LDS tile
#pragma unroll
    for (int nt = 0; nt < 4; ++nt)
#pragma unroll
      for (int r = 0; r < 8; ++r)
        sP[(wave * 16 + r + 8 * lh) * QP + nt * 16 + lm] = (bf16)s[nt][r];
    __syncthreads();

    // O += P @ V
#pragma unroll
    for (int kk = 0; kk < 2; ++kk) {
      v16bf ap = lds_load_a_frag(&sP[(wave * 16) * QP + kk * 32], QP);
#pragma unroll
      for (int nt = 0; nt < 4; ++nt) {
        v16bf bv = lds_load_b_frag(&sVt[(nt * 16) * QP + kk * 32], QP);
        o[nt] = WMMA_BF16(ap, bv, o[nt]);
      }
    }
    __syncthreads();
  }

  // Write normalized output: attnOut [B*T, nh*hd]
  bf16* obase = attnOut + (size_t)(b * T) * (nh * hd) + h * hd;
#pragma unroll
  for (int nt = 0; nt < 4; ++nt) {
#pragma unroll
    for (int r = 0; r < 8; ++r) {
      int row = q0 + wave * 16 + r + 8 * lh;
      int d   = nt * 16 + lm;
      obase[(size_t)row * (nh * hd) + d] = (bf16)(o[nt][r] / lrun[r]);
    }
  }
}

// ---------------------------------------------------------------------------
// Launch
// ---------------------------------------------------------------------------
extern "C" void kernel_launch(void* const* d_in, const int* in_sizes, int n_in,
                              void* d_out, int out_size, void* d_ws, size_t ws_size,
                              hipStream_t stream) {
  (void)in_sizes; (void)n_in; (void)out_size; (void)ws_size;

  const float* x     = (const float*)d_in[0];
  const float* qkv_w = (const float*)d_in[1];
  const float* qkv_b = (const float*)d_in[2];
  const float* out_w = (const float*)d_in[3];
  const float* out_b = (const float*)d_in[4];
  const float* ln1_g = (const float*)d_in[5];
  const float* ln1_b = (const float*)d_in[6];
  const float* ln2_g = (const float*)d_in[7];
  const float* ln2_b = (const float*)d_in[8];
  const float* fc1_w = (const float*)d_in[9];
  const float* fc1_b = (const float*)d_in[10];
  const float* fc2_w = (const float*)d_in[11];
  const float* fc2_b = (const float*)d_in[12];
  float* out = (float*)d_out;

  const int B = 2, T = 2048, C = 1024, H = 4096, NH = 16, HD = 64;
  const int M = B * T;  // 4096 token rows

  char* ws = (char*)d_ws;
  size_t off = 0;
  auto alloc = [&](size_t bytes) -> void* {
    void* p = ws + off;
    off = (off + bytes + 255) & ~(size_t)255;
    return p;
  };

  bf16*  nx    = (bf16*) alloc((size_t)M * C * 2);        // LN1(x)
  bf16*  qkvb  = (bf16*) alloc((size_t)M * 3 * C * 2);    // QKV activations
  bf16*  attnb = (bf16*) alloc((size_t)M * C * 2);        // attention output
  float* x1    = (float*)alloc((size_t)M * C * 4);        // x + attn proj
  bf16*  nx2   = (bf16*) alloc((size_t)M * C * 2);        // LN2(x1)
  bf16*  hb    = (bf16*) alloc((size_t)M * H * 2);        // GELU(fc1)
  bf16*  wqkv  = (bf16*) alloc((size_t)3 * C * C * 2);
  bf16*  wout  = (bf16*) alloc((size_t)C * C * 2);
  bf16*  wfc1  = (bf16*) alloc((size_t)H * C * 2);
  bf16*  wfc2  = (bf16*) alloc((size_t)C * H * 2);

  auto cvt = [&](const float* s, bf16* d, int n) {
    f32_to_bf16_kernel<<<(n + 255) / 256, 256, 0, stream>>>(s, d, n);
  };
  cvt(qkv_w, wqkv, 3 * C * C);
  cvt(out_w, wout, C * C);
  cvt(fc1_w, wfc1, H * C);
  cvt(fc2_w, wfc2, C * H);

  // 1) LN1 -> bf16
  layernorm_to_bf16_kernel<<<M, 256, 0, stream>>>(x, ln1_g, ln1_b, nx, C);

  // 2) QKV projection: [4096,1024] @ [3072,1024]^T -> bf16
  gemm_bf16_wmma_kernel<<<dim3(3 * C / BN, M / BM), 256, 0, stream>>>(
      nx, wqkv, qkv_b, nullptr, nullptr, qkvb, M, 3 * C, C, 0);

  // 3) Causal flash attention per (b,h,64-query block)
  flash_attn_wmma_kernel<<<dim3(T / 64, B * NH), 128, 0, stream>>>(
      qkvb, attnb, T, NH, HD);

  // 4) Output projection + residual -> x1 (f32)
  gemm_bf16_wmma_kernel<<<dim3(C / BN, M / BM), 256, 0, stream>>>(
      attnb, wout, out_b, x, x1, nullptr, M, C, C, 1);

  // 5) LN2 -> bf16
  layernorm_to_bf16_kernel<<<M, 256, 0, stream>>>(x1, ln2_g, ln2_b, nx2, C);

  // 6) FC1 + exact GELU -> bf16
  gemm_bf16_wmma_kernel<<<dim3(H / BN, M / BM), 256, 0, stream>>>(
      nx2, wfc1, fc1_b, nullptr, nullptr, hb, M, H, C, 2);

  // 7) FC2 + residual -> d_out (f32)
  gemm_bf16_wmma_kernel<<<dim3(C / BN, M / BM), 256, 0, stream>>>(
      hb, wfc2, fc2_b, x1, out, nullptr, M, C, H, 1);
}
